// Decoder_80315888435460
// MI455X (gfx1250) — matrix operand order repaired and checked by reading, not compiled, not run on hardware
//
#include <hip/hip_runtime.h>
#include <hip/hip_bf16.h>

// ---------------------------------------------------------------------------
// Tacotron2-style decoder for MI455X (gfx1250).
//  * bf16 WMMA (16x16x32) for the two LSTM gate GEMMs (M=32 -> 2 row tiles)
//  * weights converted+pre-swizzled once into WMMA B-fragment order (L2-resident)
//  * activations staged to LDS via async global->LDS copies, double buffered
//  * per-step kernels: attLSTM -> attention -> decLSTM -> projection
// ---------------------------------------------------------------------------

typedef __attribute__((ext_vector_type(16))) __bf16 v16bf;
typedef __attribute__((ext_vector_type(8)))  __bf16 v8bf;
typedef __attribute__((ext_vector_type(8)))  float  v8f;
typedef int v4i __attribute__((vector_size(16)));
typedef __attribute__((address_space(1))) v4i* gv4i_p;   // global int32x4*
typedef __attribute__((address_space(3))) v4i* lv4i_p;   // LDS    int32x4*

#define NB   32      // batch
#define TENC 400
#define DENC 512
#define TMEL 500
#define NMEL 80
#define PRE  256
#define HID  1024    // both LSTMs
#define ADIM 128
#define LOCF 32
#define KS   31

#define KCHUNK 256          // K elements staged per double-buffer chunk
#define KTPC   8            // WMMA k-tiles (32 wide) per chunk
#define APAD   (KCHUNK + 8) // padded LDS row stride (bank spread)

#if defined(__has_builtin)
# if __has_builtin(__builtin_amdgcn_global_load_async_to_lds_b128)
#  define HAVE_ASYNC_LDS 1
# endif
#endif
#ifndef HAVE_ASYNC_LDS
# define HAVE_ASYNC_LDS 0
#endif

__device__ __forceinline__ __bf16 f2bf(float f) {
    unsigned u = __builtin_bit_cast(unsigned, f);
    unsigned r = (u + 0x7FFFu + ((u >> 16) & 1u)) >> 16;
    unsigned short h = (unsigned short)r;
    return __builtin_bit_cast(__bf16, h);
}
__device__ __forceinline__ float sigm(float x) { return 1.0f / (1.0f + __expf(-x)); }

__device__ __forceinline__ void async_wait0() {
#if HAVE_ASYNC_LDS
# if __has_builtin(__builtin_amdgcn_s_wait_asynccnt)
    __builtin_amdgcn_s_wait_asynccnt(0);
# else
    asm volatile("s_wait_asynccnt 0x0" ::: "memory");
# endif
#endif
}

// --------------------------- prep kernels ----------------------------------

__global__ void zero_kernel(unsigned int* p, size_t n) {
    size_t i = (size_t)blockIdx.x * 256 + threadIdx.x;
    if (i < n) p[i] = 0u;
}

// Pack fused [W1 | W2] (row-major, N=4096 rows, Ktot cols) into WMMA
// B-fragment tiles: tile (nt,kt) stored as 512 contiguous bf16:
//   elem = lane*16 + j ; n = nt*16 + (lane&15) ; k = kt*32 + (lane>>4)*8 + (j&7) + (j>>3)*16
__global__ void swizzle_kernel(const float* __restrict__ W1, int K1,
                               const float* __restrict__ W2, int K2,
                               int Ktot, __bf16* __restrict__ out, size_t total) {
    size_t idx = (size_t)blockIdx.x * 256 + threadIdx.x;
    if (idx >= total) return;
    int within = (int)(idx & 511);
    size_t tile = idx >> 9;
    int numKT = Ktot >> 5;
    int kt = (int)(tile % numKT);
    int nt = (int)(tile / numKT);
    int lane = within >> 4, j = within & 15;
    int n = nt * 16 + (lane & 15);
    int k = kt * 32 + (lane >> 4) * 8 + (j & 7) + (j >> 3) * 16;
    float w = (k < K1) ? W1[(size_t)n * K1 + k] : W2[(size_t)n * K2 + (k - K1)];
    out[idx] = f2bf(w);
}

// p1[t,b,p] = relu(sum_m x[t,b,m] * W_pre1[p,m]); x[0]=0, x[t]=dec_in[b,m,t-1]
__global__ void prenet1_kernel(const float* __restrict__ dec_in,
                               const float* __restrict__ W_pre1,
                               float* __restrict__ p1) {
    size_t idx = (size_t)blockIdx.x * 256 + threadIdx.x;
    if (idx >= (size_t)TMEL * NB * PRE) return;
    int p = (int)(idx & 255);
    size_t tb = idx >> 8;
    int b = (int)(tb % NB);
    int t = (int)(tb / NB);
    float s = 0.0f;
    if (t > 0) {
        const float* xr = dec_in + (size_t)b * NMEL * TMEL + (t - 1);
        const float* wr = W_pre1 + (size_t)p * NMEL;
        for (int m = 0; m < NMEL; ++m) s += wr[m] * xr[(size_t)m * TMEL];
    }
    p1[idx] = fmaxf(s, 0.0f);
}

// steps_bf[t,b,q] = bf16(relu(sum_p p1[t,b,p] * W_pre2[q,p]))
__global__ void prenet2_kernel(const float* __restrict__ p1,
                               const float* __restrict__ W_pre2,
                               __bf16* __restrict__ steps_bf) {
    size_t idx = (size_t)blockIdx.x * 256 + threadIdx.x;
    if (idx >= (size_t)TMEL * NB * PRE) return;
    int q = (int)(idx & 255);
    size_t tb = idx >> 8;
    const float* pr = p1 + tb * PRE;
    const float* wr = W_pre2 + (size_t)q * PRE;
    float s = 0.0f;
    for (int p = 0; p < PRE; ++p) s += wr[p] * pr[p];
    steps_bf[idx] = f2bf(fmaxf(s, 0.0f));
}

// pmem[b,t,a] = sum_d memory[b,t,d] * W_mem[a,d]
__global__ void pmem_kernel(const float* __restrict__ memory,
                            const float* __restrict__ W_mem,
                            float* __restrict__ pmem) {
    size_t idx = (size_t)blockIdx.x * 256 + threadIdx.x;
    if (idx >= (size_t)NB * TENC * ADIM) return;
    int a = (int)(idx & 127);
    size_t bt = idx >> 7;
    const float* mr = memory + bt * DENC;
    const float* wr = W_mem + (size_t)a * DENC;
    float s = 0.0f;
    for (int d = 0; d < DENC; ++d) s += wr[d] * mr[d];
    pmem[idx] = s;
}

// ------------------------- fused LSTM (WMMA) -------------------------------

// Stage one 32 x KCHUNK bf16 activation chunk (segmented source) into LDS.
// 16-byte granules; segment boundaries are multiples of 256 so each granule
// lies in exactly one segment.
__device__ __forceinline__ void stage_chunk(
    __bf16* __restrict__ dst, int kBase,
    const __bf16* __restrict__ seg0, int len0,
    const __bf16* __restrict__ seg1, int len1,
    const __bf16* __restrict__ seg2, int len2,
    int kS1, int kS2, int tid)
{
    for (int c = tid; c < (32 * KCHUNK) / 8; c += 256) {
        int e  = c * 8;                 // bf16 element offset within chunk
        int m  = e >> 8;                // / KCHUNK
        int kl = e & (KCHUNK - 1);
        int kG = kBase + kl;
        const __bf16* g;
        if (kG < kS1)      g = seg0 + (size_t)m * len0 + kG;
        else if (kG < kS2) g = seg1 + (size_t)m * len1 + (kG - kS1);
        else               g = seg2 + (size_t)m * len2 + (kG - kS2);
        __bf16* l = dst + m * APAD + kl;
#if HAVE_ASYNC_LDS
        __builtin_amdgcn_global_load_async_to_lds_b128(
            (gv4i_p)(void*)g, (lv4i_p)(void*)l, 0, 0);
#else
        *(v8bf*)l = *(const v8bf*)g;
#endif
    }
}

// Gates = [seg0|seg1|seg2](32 x Ktot, bf16) @ Wf(Ktot x 4096, pre-swizzled) + b1 + b2
// Grid: 32 workgroups, each owns 32 columns of every gate block. 8 waves:
// wave = (gate<<1)|mTile; each wave accumulates two 16x16 f32 tiles.
__global__ __launch_bounds__(256) void lstm_wmma_kernel(
    const __bf16* __restrict__ Wf, int Ktot,
    const __bf16* __restrict__ seg0, int len0,
    const __bf16* __restrict__ seg1, int len1,
    const __bf16* __restrict__ seg2, int len2,
    const float* __restrict__ b1, const float* __restrict__ b2,
    float* __restrict__ c_state, float* __restrict__ h_state,
    __bf16* __restrict__ h_bf)
{
    __shared__ __bf16 sA[2][32 * APAD];  // double-buffered activations (~33 KB)
    __shared__ float  sG[4 * 32 * 32];   // [gate][row][col] gate exchange (16 KB)
    const int tid  = threadIdx.x;
    const int wave = tid >> 5;
    const int lane = tid & 31;
    const int wg   = blockIdx.x;         // 0..31
    const int mTile = wave & 1;
    const int gate  = wave >> 1;
    const int numKT = Ktot >> 5;
    const int numCh = Ktot >> 8;
    const int kS1 = len0, kS2 = len0 + len1;

    stage_chunk(sA[0], 0, seg0, len0, seg1, len1, seg2, len2, kS1, kS2, tid);

    v8f acc[2] = {};
    const int ntBase = ((gate * HID + wg * 32) >> 4);
    const int mRow   = mTile * 16 + (lane & 15);
    const int aOff   = (lane >> 4) * 8;  // A frag: K 0..7|16..23 vs 8..15|24..31

    for (int ch = 0; ch < numCh; ++ch) {
        const int buf = ch & 1;
        async_wait0();
        __syncthreads();                 // chunk `ch` visible; prev buffer free
        if (ch + 1 < numCh) {
            stage_chunk(sA[buf ^ 1], (ch + 1) * KCHUNK,
                        seg0, len0, seg1, len1, seg2, len2, kS1, kS2, tid);
            __builtin_prefetch(Wf + ((size_t)ntBase * numKT + (ch + 1) * KTPC) * 512 + lane * 16, 0, 2);
        }
        const __bf16* arow = &sA[buf][mRow * APAD];
#pragma unroll
        for (int k0 = 0; k0 < KTPC; ++k0) {
            const int kt = ch * KTPC + k0;
            v8bf alo = *(const v8bf*)(arow + k0 * 32 + aOff);
            v8bf ahi = *(const v8bf*)(arow + k0 * 32 + aOff + 16);
            v16bf afrag;
#pragma unroll
            for (int i = 0; i < 8; ++i) { afrag[i] = alo[i]; afrag[i + 8] = ahi[i]; }
            const __bf16* bp0 = Wf + ((size_t)ntBase * numKT + kt) * 512 + lane * 16;
            const __bf16* bp1 = Wf + ((size_t)(ntBase + 1) * numKT + kt) * 512 + lane * 16;
            v16bf bf0 = *(const v16bf*)bp0;
            v16bf bf1 = *(const v16bf*)bp1;
            acc[0] = __builtin_amdgcn_wmma_f32_16x16x32_bf16(
                false, afrag, false, bf0, (short)0, acc[0], false, false);
            acc[1] = __builtin_amdgcn_wmma_f32_16x16x32_bf16(
                false, afrag, false, bf1, (short)0, acc[1], false, false);
        }
    }

    // D layout: lane holds col n=lane&15; VGPR r holds row 8*(lane>>4)+r
#pragma unroll
    for (int i = 0; i < 2; ++i)
#pragma unroll
        for (int r = 0; r < 8; ++r) {
            int mm  = mTile * 16 + (lane >> 4) * 8 + r;
            int col = i * 16 + (lane & 15);
            sG[(gate * 32 + mm) * 32 + col] = acc[i][r];
        }
    __syncthreads();

    for (int idx = tid; idx < 32 * 32; idx += 256) {
        int mm = idx >> 5, col = idx & 31;
        int n  = wg * 32 + col;
        float gi = sG[(0 * 32 + mm) * 32 + col] + b1[n]            + b2[n];
        float gf = sG[(1 * 32 + mm) * 32 + col] + b1[HID + n]      + b2[HID + n];
        float gg = sG[(2 * 32 + mm) * 32 + col] + b1[2 * HID + n]  + b2[2 * HID + n];
        float go = sG[(3 * 32 + mm) * 32 + col] + b1[3 * HID + n]  + b2[3 * HID + n];
        float cn = sigm(gf) * c_state[mm * HID + n] + sigm(gi) * tanhf(gg);
        float h  = sigm(go) * tanhf(cn);
        c_state[mm * HID + n] = cn;
        h_state[mm * HID + n] = h;
        h_bf[mm * HID + n]    = f2bf(h);
    }
}

// --------------------------- attention step --------------------------------
// one workgroup per batch item; conv + energies + masked softmax + context
__global__ __launch_bounds__(512) void attention_kernel(
    const float* __restrict__ h_a, const float* __restrict__ W_q,
    const float* __restrict__ W_conv, const float* __restrict__ W_loc,
    const float* __restrict__ vvec, const float* __restrict__ pmem,
    const float* __restrict__ memory, const int* __restrict__ lens,
    float* __restrict__ aw, float* __restrict__ awc,
    float* __restrict__ ctx, __bf16* __restrict__ ctx_bf,
    float* __restrict__ aligns_out, int t_step)
{
    __shared__ float s_paw[TENC + 32], s_pawc[TENC + 32];
    __shared__ float s_q[ADIM];
    __shared__ float s_locraw[LOCF * TENC];
    __shared__ float s_e[TENC];
    __shared__ float s_red[512];
    const int b = blockIdx.x, tid = threadIdx.x;

    for (int p = tid; p < TENC + 32; p += 512) {
        int te = p - 16;
        bool ok = (te >= 0) && (te < TENC);
        s_paw[p]  = ok ? aw[b * TENC + te]  : 0.0f;
        s_pawc[p] = ok ? awc[b * TENC + te] : 0.0f;
    }
    if (tid < ADIM) {
        const float* hp = h_a + (size_t)b * HID;
        const float* wp = W_q + (size_t)tid * HID;
        float s = 0.0f;
        for (int k = 0; k < HID; ++k) s += hp[k] * wp[k];
        s_q[tid] = s;
    }
    __syncthreads();

    for (int idx = tid; idx < LOCF * TENC; idx += 512) {
        int f = idx / TENC, te = idx % TENC;
        const float* w0 = W_conv + f * 2 * KS;
        const float* w1 = w0 + KS;
        float s = 0.0f;
        for (int k = 0; k < KS; ++k)
            s += w0[k] * s_paw[te + k + 1] + w1[k] * s_pawc[te + k + 1];
        s_locraw[f * TENC + te] = s;
    }
    __syncthreads();

    for (int te = tid; te < TENC; te += 512) {
        const float* pm = pmem + ((size_t)b * TENC + te) * ADIM;
        float e = 0.0f;
        for (int a = 0; a < ADIM; ++a) {
            const float* wl = W_loc + a * LOCF;
            float la = 0.0f;
#pragma unroll
            for (int f = 0; f < LOCF; ++f) la += wl[f] * s_locraw[f * TENC + te];
            e += vvec[a] * tanhf(s_q[a] + la + pm[a]);
        }
        if (te >= lens[b]) e = -1e9f;
        s_e[te] = e;
    }
    __syncthreads();

    float mx = -1e30f;
    for (int te = tid; te < TENC; te += 512) mx = fmaxf(mx, s_e[te]);
    s_red[tid] = mx; __syncthreads();
    for (int s = 256; s > 0; s >>= 1) {
        if (tid < s) s_red[tid] = fmaxf(s_red[tid], s_red[tid + s]);
        __syncthreads();
    }
    mx = s_red[0]; __syncthreads();

    float sum = 0.0f;
    for (int te = tid; te < TENC; te += 512) {
        float ex = __expf(s_e[te] - mx);
        s_e[te] = ex; sum += ex;
    }
    s_red[tid] = sum; __syncthreads();
    for (int s = 256; s > 0; s >>= 1) {
        if (tid < s) s_red[tid] += s_red[tid + s];
        __syncthreads();
    }
    float inv = 1.0f / s_red[0]; __syncthreads();

    for (int te = tid; te < TENC; te += 512) {
        float awn = s_e[te] * inv;
        s_e[te] = awn;
        aw[b * TENC + te]  = awn;
        awc[b * TENC + te] = s_pawc[te + 16] + awn;
        aligns_out[(size_t)b * TMEL * TENC + (size_t)t_step * TENC + te] = awn;
    }
    __syncthreads();

    {   // ctx[b,d] = sum_t aw_n[t] * memory[b,t,d]
        int d = tid;
        const float* mp = memory + (size_t)b * TENC * DENC + d;
        float s = 0.0f;
        for (int te = 0; te < TENC; ++te) s += s_e[te] * mp[(size_t)te * DENC];
        ctx[b * DENC + d]    = s;
        ctx_bf[b * DENC + d] = f2bf(s);
    }
}

// --------------------------- projection ------------------------------------
__global__ __launch_bounds__(128) void proj_kernel(
    const float* __restrict__ h_d, const float* __restrict__ ctx,
    const float* __restrict__ W_proj, const float* __restrict__ b_proj,
    const float* __restrict__ W_stop, const float* __restrict__ b_stop,
    float* __restrict__ out, int t_step)
{
    int b = blockIdx.x, j = threadIdx.x;
    if (j > NMEL) return;  // 0..79 mel, 80 stop
    const float* w = (j < NMEL) ? (W_proj + (size_t)j * (HID + DENC)) : W_stop;
    float s = (j < NMEL) ? b_proj[j] : b_stop[0];
    const float* hp = h_d + (size_t)b * HID;
    const float* cp = ctx + (size_t)b * DENC;
    for (int k = 0; k < HID; ++k)  s += w[k] * hp[k];
    for (int k = 0; k < DENC; ++k) s += w[HID + k] * cp[k];
    if (j < NMEL) out[(size_t)b * NMEL * TMEL + (size_t)j * TMEL + t_step] = s;
    else          out[(size_t)NB * NMEL * TMEL + (size_t)b * TMEL + t_step] = s;
}

// ---------------------------------------------------------------------------

extern "C" void kernel_launch(void* const* d_in, const int* in_sizes, int n_in,
                              void* d_out, int out_size, void* d_ws, size_t ws_size,
                              hipStream_t stream) {
    // --- input index resolution (insertion vs JAX-sorted flattening) ---
    // A: full insertion order; B: outer insertion + params sorted; C: fully sorted
    int I[22];
    if (n_in >= 1 && in_sizes[0] == 1280000) {           // scheme C
        int c[22] = {2,0,4,16,3,15,7,18,6,17,10,11,13,9,5,8,21,12,19,14,20,1};
        for (int i = 0; i < 22; ++i) I[i] = c[i];
    } else if (n_in >= 3 && in_sizes[2] == 3145728) {    // scheme A
        for (int i = 0; i < 22; ++i) I[i] = i;
    } else {                                             // scheme B
        int c[22] = {0,1,3,15,2,14,6,17,5,16,9,10,12,8,4,7,20,11,18,13,19,21};
        for (int i = 0; i < 22; ++i) I[i] = c[i];
    }
    const float* memory = (const float*)d_in[I[0]];
    const float* dec_in = (const float*)d_in[I[1]];
    const float* W_aih  = (const float*)d_in[I[2]];
    const float* b_aih  = (const float*)d_in[I[3]];
    const float* W_ahh  = (const float*)d_in[I[4]];
    const float* b_ahh  = (const float*)d_in[I[5]];
    const float* W_dih  = (const float*)d_in[I[6]];
    const float* b_dih  = (const float*)d_in[I[7]];
    const float* W_dhh  = (const float*)d_in[I[8]];
    const float* b_dhh  = (const float*)d_in[I[9]];
    const float* W_pre1 = (const float*)d_in[I[10]];
    const float* W_pre2 = (const float*)d_in[I[11]];
    const float* W_q    = (const float*)d_in[I[12]];
    const float* W_mem  = (const float*)d_in[I[13]];
    const float* W_conv = (const float*)d_in[I[14]];
    const float* W_loc  = (const float*)d_in[I[15]];
    const float* vvec   = (const float*)d_in[I[16]];
    const float* W_proj = (const float*)d_in[I[17]];
    const float* b_proj = (const float*)d_in[I[18]];
    const float* W_stop = (const float*)d_in[I[19]];
    const float* b_stop = (const float*)d_in[I[20]];
    const int*   lens   = (const int*)  d_in[I[21]];

    // --- workspace carve-out (~68 MB; all bf16 weights stay L2-resident) ---
    char* ws = (char*)d_ws;
    size_t off = 0;
    auto carve = [&](size_t bytes) { size_t o = off; off += (bytes + 255) & ~(size_t)255; return o; };
    const int KA = PRE + DENC + HID;       // 1792
    const int KD = HID + DENC + HID;       // 2560
    const size_t nWa = (size_t)4 * HID * KA, nWd = (size_t)4 * HID * KD;
    __bf16* Wa_f    = (__bf16*)(ws + carve(nWa * 2));
    __bf16* Wd_f    = (__bf16*)(ws + carve(nWd * 2));
    __bf16* steps_bf= (__bf16*)(ws + carve((size_t)TMEL * NB * PRE * 2));
    float*  p1      = (float*) (ws + carve((size_t)TMEL * NB * PRE * 4));
    float*  pmem    = (float*) (ws + carve((size_t)NB * TENC * ADIM * 4));
    size_t stateOff = off;
    float*  h_a     = (float*) (ws + carve(NB * HID * 4));
    float*  c_a     = (float*) (ws + carve(NB * HID * 4));
    float*  h_d     = (float*) (ws + carve(NB * HID * 4));
    float*  c_d     = (float*) (ws + carve(NB * HID * 4));
    float*  ctx     = (float*) (ws + carve(NB * DENC * 4));
    float*  aw      = (float*) (ws + carve(NB * TENC * 4));
    float*  awc     = (float*) (ws + carve(NB * TENC * 4));
    __bf16* h_a_bf  = (__bf16*)(ws + carve(NB * HID * 2));
    __bf16* h_d_bf  = (__bf16*)(ws + carve(NB * HID * 2));
    __bf16* ctx_bf  = (__bf16*)(ws + carve(NB * DENC * 2));
    size_t stateBytes = off - stateOff;
    (void)ws_size;

    float* out = (float*)d_out;
    float* aligns_out = out + (size_t)NB * NMEL * TMEL + (size_t)NB * TMEL;

    // --- prep ---
    {
        size_t nz = stateBytes / 4;
        zero_kernel<<<dim3((unsigned)((nz + 255) / 256)), dim3(256), 0, stream>>>(
            (unsigned int*)(ws + stateOff), nz);
        swizzle_kernel<<<dim3((unsigned)((nWa + 255) / 256)), dim3(256), 0, stream>>>(
            W_aih, PRE + DENC, W_ahh, HID, KA, Wa_f, nWa);
        swizzle_kernel<<<dim3((unsigned)((nWd + 255) / 256)), dim3(256), 0, stream>>>(
            W_dih, HID + DENC, W_dhh, HID, KD, Wd_f, nWd);
        size_t np = (size_t)TMEL * NB * PRE;
        prenet1_kernel<<<dim3((unsigned)((np + 255) / 256)), dim3(256), 0, stream>>>(dec_in, W_pre1, p1);
        prenet2_kernel<<<dim3((unsigned)((np + 255) / 256)), dim3(256), 0, stream>>>(p1, W_pre2, steps_bf);
        size_t nm = (size_t)NB * TENC * ADIM;
        pmem_kernel<<<dim3((unsigned)((nm + 255) / 256)), dim3(256), 0, stream>>>(memory, W_mem, pmem);
    }

    // --- sequential decode: 500 steps, 4 kernels each (graph-capturable) ---
    for (int t = 0; t < TMEL; ++t) {
        const __bf16* xt = steps_bf + (size_t)t * NB * PRE;
        lstm_wmma_kernel<<<dim3(32), dim3(256), 0, stream>>>(
            Wa_f, KA, xt, PRE, ctx_bf, DENC, h_a_bf, HID,
            b_aih, b_ahh, c_a, h_a, h_a_bf);
        attention_kernel<<<dim3(NB), dim3(512), 0, stream>>>(
            h_a, W_q, W_conv, W_loc, vvec, pmem, memory, lens,
            aw, awc, ctx, ctx_bf, aligns_out, t);
        lstm_wmma_kernel<<<dim3(32), dim3(256), 0, stream>>>(
            Wd_f, KD, h_a_bf, HID, ctx_bf, DENC, h_d_bf, HID,
            b_dih, b_dhh, c_d, h_d, h_d_bf);
        proj_kernel<<<dim3(NB), dim3(128), 0, stream>>>(
            h_d, ctx, W_proj, b_proj, W_stop, b_stop, out, t);
    }
}
